// CosineAttention_85212151152953
// MI455X (gfx1250) — matrix operand order repaired and checked by reading, not compiled, not run on hardware
//
#include <hip/hip_runtime.h>
#include <hip/hip_bf16.h>
#include <math.h>

typedef __attribute__((ext_vector_type(16))) _Float16 v16h;
typedef __attribute__((ext_vector_type(8)))  _Float16 v8h;
typedef __attribute__((ext_vector_type(8)))  float    v8f;

#define HEADS 16
#define DHEAD 64
#define DIM   1024

__device__ __forceinline__ v8f vzero8() {
  v8f z;
#pragma unroll
  for (int i = 0; i < 8; ++i) z[i] = 0.0f;
  return z;
}

// A fragment (16x32 f16). Lane l=lane&15 holds row M=l; half hf=lane>>4.
// VGPR0..7 hold K = hf*16 + {0..15} (contiguous).
// Caller passes p = &A[row*ld + kbase + hf*16].
__device__ __forceinline__ v16h load_fragA(const _Float16* p) {
  v8h lo = *(const v8h*)(p);
  v8h hi = *(const v8h*)(p + 8);
  v16h r;
#pragma unroll
  for (int i = 0; i < 8; ++i) { r[i] = lo[i]; r[8 + i] = hi[i]; }
  return r;
}

// B fragment (32x16 f16). Lane l holds column N=l; lanes 0-15 K=0..15,
// lanes 16-31 K=16..31 (contiguous). Caller passes p = &BT[n*ld + kbase + hf*16].
__device__ __forceinline__ v16h load_fragB(const _Float16* p) {
  v8h lo = *(const v8h*)(p);
  v8h hi = *(const v8h*)(p + 8);
  v16h r;
#pragma unroll
  for (int i = 0; i < 8; ++i) { r[i] = lo[i]; r[8 + i] = hi[i]; }
  return r;
}

__device__ __forceinline__ v8f wmma16(v16h a, v16h b, v8f c) {
  return __builtin_amdgcn_wmma_f32_16x16x32_f16(false, a, false, b, (short)0, c,
                                                false, false);
}

__device__ __forceinline__ float waveReduceSum(float v) {
#pragma unroll
  for (int off = 16; off > 0; off >>= 1) v += __shfl_xor(v, off, 32);
  return v;
}

// ---------------- LayerNorm (row of 1024) -> f16 ----------------
__global__ __launch_bounds__(256) void k_ln(const float* __restrict__ X,
                                            _Float16* __restrict__ O,
                                            const float* __restrict__ gamma,
                                            const float* __restrict__ beta) {
  __shared__ float sh[2][8];
  __shared__ float stats[2];
  int row = blockIdx.x;
  const float* xr = X + (size_t)row * DIM;
  float v[4], s = 0.f, ss = 0.f;
#pragma unroll
  for (int i = 0; i < 4; ++i) {
    v[i] = xr[threadIdx.x + 256 * i];
    s += v[i];
    ss += v[i] * v[i];
  }
  s = waveReduceSum(s);
  ss = waveReduceSum(ss);
  int wave = threadIdx.x >> 5, lane = threadIdx.x & 31;
  if (lane == 0) { sh[0][wave] = s; sh[1][wave] = ss; }
  __syncthreads();
  if (threadIdx.x == 0) {
    float S = 0.f, SS = 0.f;
#pragma unroll
    for (int i = 0; i < 8; ++i) { S += sh[0][i]; SS += sh[1][i]; }
    float mu = S * (1.0f / DIM);
    float var = SS * (1.0f / DIM) - mu * mu;
    stats[0] = mu;
    stats[1] = rsqrtf(var + 1e-5f);
  }
  __syncthreads();
  float mu = stats[0], rstd = stats[1];
  _Float16* orow = O + (size_t)row * DIM;
#pragma unroll
  for (int i = 0; i < 4; ++i) {
    int c = threadIdx.x + 256 * i;
    orow[c] = (_Float16)(((v[i] - mu) * rstd) * gamma[c] + beta[c]);
  }
}

// ---------------- transpose + f16 convert of a 1024x1024 weight ----------------
__global__ __launch_bounds__(256) void k_transpose_h(const float* __restrict__ W,
                                                     _Float16* __restrict__ WT) {
  int c = blockIdx.x;
  for (int k = threadIdx.x; k < DIM; k += blockDim.x)
    WT[(size_t)c * DIM + k] = (_Float16)W[(size_t)k * DIM + c];
}

// ====== projection GEMM, fused per-head normalize/pack epilogue ======
// C = XH @ W_in over all 20480 rows (q:0..4095, k:4096..12287, v:12288..20479).
// Wave tile 32x64 (2 A-frags x 4 B-frags, 8 acc), software-pipelined loads.
// Block = 8 waves (4 M x 2 N) -> 128x128 tile. grid = (1024/128, 20480/128).
// Epilogue: wave's 64-wide N-strip == one head; per-row L2 norm over d=64 is
// computed in-register (4-acc sum of squares + 16-lane shfl reduce), then:
//   q rows -> QH row-major (normalized), k rows -> KT[b,h,d,m] (normalized),
//   v rows -> VT[b,h,d,m].
__global__ __launch_bounds__(256) void k_gemm_proj(
    const _Float16* __restrict__ A, const _Float16* __restrict__ BT,
    _Float16* __restrict__ QH, _Float16* __restrict__ KT,
    _Float16* __restrict__ VT) {
  const int K = DIM;
  int lane = threadIdx.x & 31;
  int wave = threadIdx.x >> 5;
  int wm = wave & 3, wn = wave >> 2;
  int l = lane & 15, hf = lane >> 4;
  int row0 = blockIdx.y * 128 + wm * 32;
  int col0 = blockIdx.x * 128 + wn * 64;

  const _Float16* a0p = A + (size_t)(row0 + l) * K + hf * 16;
  const _Float16* a1p = a0p + (size_t)16 * K;
  const _Float16* b0p = BT + (size_t)(col0 + 0 * 16 + l) * K + hf * 16;
  const _Float16* b1p = BT + (size_t)(col0 + 1 * 16 + l) * K + hf * 16;
  const _Float16* b2p = BT + (size_t)(col0 + 2 * 16 + l) * K + hf * 16;
  const _Float16* b3p = BT + (size_t)(col0 + 3 * 16 + l) * K + hf * 16;

  v8f acc[8];
#pragma unroll
  for (int i = 0; i < 8; ++i) acc[i] = vzero8();

  v16h a0 = load_fragA(a0p), a1 = load_fragA(a1p);
  v16h b0 = load_fragB(b0p), b1 = load_fragB(b1p);
  v16h b2 = load_fragB(b2p), b3 = load_fragB(b3p);

  for (int k = 32; k < K; k += 32) {
    v16h na0 = load_fragA(a0p + k), na1 = load_fragA(a1p + k);
    v16h nb0 = load_fragB(b0p + k), nb1 = load_fragB(b1p + k);
    v16h nb2 = load_fragB(b2p + k), nb3 = load_fragB(b3p + k);
    acc[0] = wmma16(a0, b0, acc[0]);
    acc[1] = wmma16(a0, b1, acc[1]);
    acc[2] = wmma16(a0, b2, acc[2]);
    acc[3] = wmma16(a0, b3, acc[3]);
    acc[4] = wmma16(a1, b0, acc[4]);
    acc[5] = wmma16(a1, b1, acc[5]);
    acc[6] = wmma16(a1, b2, acc[6]);
    acc[7] = wmma16(a1, b3, acc[7]);
    a0 = na0; a1 = na1; b0 = nb0; b1 = nb1; b2 = nb2; b3 = nb3;
  }
  acc[0] = wmma16(a0, b0, acc[0]);
  acc[1] = wmma16(a0, b1, acc[1]);
  acc[2] = wmma16(a0, b2, acc[2]);
  acc[3] = wmma16(a0, b3, acc[3]);
  acc[4] = wmma16(a1, b0, acc[4]);
  acc[5] = wmma16(a1, b1, acc[5]);
  acc[6] = wmma16(a1, b2, acc[6]);
  acc[7] = wmma16(a1, b3, acc[7]);

  int h = col0 >> 6;  // head index (col0 is 64-aligned, < 1024)
#pragma unroll
  for (int im = 0; im < 2; ++im) {
    // per-row 1/(||.||+eps) over this head's 64 columns
    float sc[8];
#pragma unroll
    for (int r = 0; r < 8; ++r) {
      float s = 0.f;
#pragma unroll
      for (int j = 0; j < 4; ++j) {
        float x = acc[im * 4 + j][r];
        s += x * x;
      }
#pragma unroll
      for (int off = 1; off < 16; off <<= 1) s += __shfl_xor(s, off, 32);
      sc[r] = 1.0f / (sqrtf(s) + 1e-8f);
    }
#pragma unroll
    for (int j = 0; j < 4; ++j) {
#pragma unroll
      for (int r = 0; r < 8; ++r) {
        int M = row0 + im * 16 + hf * 8 + r;  // global row in [0, 20480)
        int c = col0 + j * 16 + l;            // 0..1023
        float val = acc[im * 4 + j][r];
        if (row0 < 4096) {  // q segment (uniform per block)
          QH[(size_t)M * DIM + c] = (_Float16)(val * sc[r]);
        } else if (row0 < 12288) {  // k segment
          int rk = M - 4096;
          int b = rk >> 11, m = rk & 2047;
          KT[((size_t)(b * HEADS + h) * DHEAD + (c & 63)) * 2048 + m] =
              (_Float16)(val * sc[r]);
        } else {  // v segment
          int rv = M - 12288;
          int b = rv >> 11, m = rv & 2047;
          VT[((size_t)(b * HEADS + h) * DHEAD + (c & 63)) * 2048 + m] =
              (_Float16)val;
        }
      }
    }
  }
}

// ---------------- S = K'^T V per head: [64x64], K-dim = 2048 ------------------
// grid = 64 (b*h); wave w -> row-tile w&3, col-tiles (w>>2)*2 + {0,1};
// double-buffered. Stores S transposed: ST[bh][n*64 + m] (B operand of Q'@S).
__global__ __launch_bounds__(256) void k_sgemm(const _Float16* __restrict__ KT,
                                               const _Float16* __restrict__ VT,
                                               _Float16* __restrict__ ST) {
  int bh = blockIdx.x;
  const _Float16* Kh = KT + (size_t)bh * DHEAD * 2048;
  const _Float16* Vh = VT + (size_t)bh * DHEAD * 2048;
  int lane = threadIdx.x & 31, wave = threadIdx.x >> 5;
  int l = lane & 15, hf = lane >> 4;
  int Mi = wave & 3, N0 = (wave >> 2) * 2;
  const _Float16* ap = Kh + (size_t)(Mi * 16 + l) * 2048 + hf * 16;
  const _Float16* b0p = Vh + (size_t)(N0 * 16 + l) * 2048 + hf * 16;
  const _Float16* b1p = b0p + (size_t)16 * 2048;
  v8f acc0 = vzero8(), acc1 = vzero8();
  v16h a = load_fragA(ap), x0 = load_fragB(b0p), x1 = load_fragB(b1p);
  for (int k = 32; k < 2048; k += 32) {
    v16h na = load_fragA(ap + k);
    v16h n0 = load_fragB(b0p + k);
    v16h n1 = load_fragB(b1p + k);
    acc0 = wmma16(a, x0, acc0);
    acc1 = wmma16(a, x1, acc1);
    a = na; x0 = n0; x1 = n1;
  }
  acc0 = wmma16(a, x0, acc0);
  acc1 = wmma16(a, x1, acc1);
#pragma unroll
  for (int r = 0; r < 8; ++r) {
    int M = Mi * 16 + hf * 8 + r;
    ST[(size_t)bh * 4096 + (size_t)(N0 * 16 + l) * 64 + M] = (_Float16)acc0[r];
    ST[(size_t)bh * 4096 + (size_t)((N0 + 1) * 16 + l) * 64 + M] =
        (_Float16)acc1[r];
  }
}

// ---------------- G = Q' @ S per head (K=64) -> f16 merged-head layout --------
// grid = (16 heads, 4096/128); wave -> 16 rows x full 64-wide head; K fully
// unrolled (2 steps), all fragments loaded up front.
__global__ __launch_bounds__(256) void k_ggemm(const _Float16* __restrict__ QH,
                                               const _Float16* __restrict__ ST,
                                               _Float16* __restrict__ GH) {
  int h = blockIdx.x;
  int lane = threadIdx.x & 31, wave = threadIdx.x >> 5;
  int l = lane & 15, hf = lane >> 4;
  int row = blockIdx.y * 128 + wave * 16 + l;
  int b = row >> 10;
  const _Float16* arow = QH + (size_t)row * DIM + h * DHEAD + hf * 16;
  const _Float16* S = ST + (size_t)(b * HEADS + h) * 4096;
  v16h a0 = load_fragA(arow);
  v16h a1 = load_fragA(arow + 32);
  v16h s0[4], s1[4];
#pragma unroll
  for (int j = 0; j < 4; ++j) {
    s0[j] = load_fragB(S + (size_t)(j * 16 + l) * 64 + hf * 16);
    s1[j] = load_fragB(S + (size_t)(j * 16 + l) * 64 + hf * 16 + 32);
  }
  v8f acc[4] = {vzero8(), vzero8(), vzero8(), vzero8()};
#pragma unroll
  for (int j = 0; j < 4; ++j) acc[j] = wmma16(a0, s0[j], acc[j]);
#pragma unroll
  for (int j = 0; j < 4; ++j) acc[j] = wmma16(a1, s1[j], acc[j]);
  int mbase = blockIdx.y * 128 + wave * 16 + hf * 8;
#pragma unroll
  for (int j = 0; j < 4; ++j)
#pragma unroll
    for (int r = 0; r < 8; ++r)
      GH[(size_t)(mbase + r) * DIM + h * DHEAD + j * 16 + l] =
          (_Float16)acc[j][r];
}

// ---------------- out = GH @ W_out + b_out (f32 output) -----------------------
// Same 32x64-per-wave double-buffered structure. grid = (1024/128, 4096/128).
__global__ __launch_bounds__(256) void k_gemm_out(
    const _Float16* __restrict__ A, const _Float16* __restrict__ BT,
    float* __restrict__ C, const float* __restrict__ bias) {
  const int K = DIM, N = DIM;
  int lane = threadIdx.x & 31;
  int wave = threadIdx.x >> 5;
  int wm = wave & 3, wn = wave >> 2;
  int l = lane & 15, hf = lane >> 4;
  int row0 = blockIdx.y * 128 + wm * 32;
  int col0 = blockIdx.x * 128 + wn * 64;

  const _Float16* a0p = A + (size_t)(row0 + l) * K + hf * 16;
  const _Float16* a1p = a0p + (size_t)16 * K;
  const _Float16* b0p = BT + (size_t)(col0 + 0 * 16 + l) * K + hf * 16;
  const _Float16* b1p = BT + (size_t)(col0 + 1 * 16 + l) * K + hf * 16;
  const _Float16* b2p = BT + (size_t)(col0 + 2 * 16 + l) * K + hf * 16;
  const _Float16* b3p = BT + (size_t)(col0 + 3 * 16 + l) * K + hf * 16;

  v8f acc[8];
#pragma unroll
  for (int i = 0; i < 8; ++i) acc[i] = vzero8();

  v16h a0 = load_fragA(a0p), a1 = load_fragA(a1p);
  v16h b0 = load_fragB(b0p), b1 = load_fragB(b1p);
  v16h b2 = load_fragB(b2p), b3 = load_fragB(b3p);

  for (int k = 32; k < K; k += 32) {
    v16h na0 = load_fragA(a0p + k), na1 = load_fragA(a1p + k);
    v16h nb0 = load_fragB(b0p + k), nb1 = load_fragB(b1p + k);
    v16h nb2 = load_fragB(b2p + k), nb3 = load_fragB(b3p + k);
    acc[0] = wmma16(a0, b0, acc[0]);
    acc[1] = wmma16(a0, b1, acc[1]);
    acc[2] = wmma16(a0, b2, acc[2]);
    acc[3] = wmma16(a0, b3, acc[3]);
    acc[4] = wmma16(a1, b0, acc[4]);
    acc[5] = wmma16(a1, b1, acc[5]);
    acc[6] = wmma16(a1, b2, acc[6]);
    acc[7] = wmma16(a1, b3, acc[7]);
    a0 = na0; a1 = na1; b0 = nb0; b1 = nb1; b2 = nb2; b3 = nb3;
  }
  acc[0] = wmma16(a0, b0, acc[0]);
  acc[1] = wmma16(a0, b1, acc[1]);
  acc[2] = wmma16(a0, b2, acc[2]);
  acc[3] = wmma16(a0, b3, acc[3]);
  acc[4] = wmma16(a1, b0, acc[4]);
  acc[5] = wmma16(a1, b1, acc[5]);
  acc[6] = wmma16(a1, b2, acc[6]);
  acc[7] = wmma16(a1, b3, acc[7]);

#pragma unroll
  for (int im = 0; im < 2; ++im) {
    int mbase = row0 + im * 16 + hf * 8;
#pragma unroll
    for (int j = 0; j < 4; ++j) {
      int c = col0 + j * 16 + l;
      float bv = bias[c];
#pragma unroll
      for (int r = 0; r < 8; ++r)
        C[(size_t)(mbase + r) * N + c] = acc[im * 4 + j][r] + bv;
    }
  }
}

extern "C" void kernel_launch(void* const* d_in, const int* in_sizes, int n_in,
                              void* d_out, int out_size, void* d_ws, size_t ws_size,
                              hipStream_t stream) {
  const float* q        = (const float*)d_in[0];
  const float* k        = (const float*)d_in[1];
  const float* v        = (const float*)d_in[2];
  const float* ln_gamma = (const float*)d_in[3];
  const float* ln_beta  = (const float*)d_in[4];
  const float* W_in     = (const float*)d_in[5];
  const float* W_out    = (const float*)d_in[6];
  const float* b_out    = (const float*)d_in[7];
  float* out = (float*)d_out;
  (void)in_sizes; (void)n_in; (void)out_size; (void)ws_size;

  const size_t QROWS = 4096, KROWS = 8192, VROWS = 8192;
  const size_t ROWS = QROWS + KROWS + VROWS;  // 20480

  char* ws = (char*)d_ws;
  size_t off = 0;
  auto alloc = [&](size_t bytes) -> void* {
    void* p = ws + off;
    off += (bytes + 255) & ~(size_t)255;
    return p;
  };
  _Float16* XH    = (_Float16*)alloc(ROWS * DIM * sizeof(_Float16));
  _Float16* WinT  = (_Float16*)alloc((size_t)DIM * DIM * sizeof(_Float16));
  _Float16* WoutT = (_Float16*)alloc((size_t)DIM * DIM * sizeof(_Float16));
  _Float16* QH    = (_Float16*)alloc(QROWS * DIM * sizeof(_Float16));
  _Float16* KT    = (_Float16*)alloc(KROWS * DIM * sizeof(_Float16));
  _Float16* VT    = (_Float16*)alloc(VROWS * DIM * sizeof(_Float16));
  _Float16* ST    = (_Float16*)alloc((size_t)64 * 64 * 64 * sizeof(_Float16));
  _Float16* GH    = (_Float16*)alloc(QROWS * DIM * sizeof(_Float16));

  dim3 blk(256);
  // 1) LayerNorm -> f16 (q, k, v stacked in XH)
  k_ln<<<dim3((unsigned)QROWS), blk, 0, stream>>>(q, XH, ln_gamma, ln_beta);
  k_ln<<<dim3((unsigned)KROWS), blk, 0, stream>>>(k, XH + QROWS * DIM, ln_gamma, ln_beta);
  k_ln<<<dim3((unsigned)VROWS), blk, 0, stream>>>(v, XH + (QROWS + KROWS) * DIM, ln_gamma, ln_beta);
  // 2) weight transposes -> f16
  k_transpose_h<<<dim3(DIM), blk, 0, stream>>>(W_in, WinT);
  k_transpose_h<<<dim3(DIM), blk, 0, stream>>>(W_out, WoutT);
  // 3) projections + fused per-head normalize/pack (no f32 F round-trip)
  k_gemm_proj<<<dim3(DIM / 128, (unsigned)(ROWS / 128)), blk, 0, stream>>>(
      XH, WinT, QH, KT, VT);
  // 4) S = K'^T V per head (associativity: (Q'K'^T)V == Q'(K'^T V))
  k_sgemm<<<dim3(64), blk, 0, stream>>>(KT, VT, ST);
  // 5) G = Q' @ S per head -> merged-head f16
  k_ggemm<<<dim3(HEADS, (unsigned)(QROWS / 128)), blk, 0, stream>>>(QH, ST, GH);
  // 6) out = G @ W_out + b_out
  k_gemm_out<<<dim3(DIM / 128, (unsigned)(QROWS / 128)), blk, 0, stream>>>(
      GH, WoutT, out, b_out);
}
